// GemmaForCausalLM_47631187313233
// MI455X (gfx1250) — compile-verified, matrix operand order checked by reading, not run on hardware
//
#include <hip/hip_runtime.h>
#include <hip/hip_bf16.h>

// ---------------------------------------------------------------------------
// Gemma block for MI455X (gfx1250, wave32). All GEMMs via v_wmma_f32_16x16x32_bf16.
// Weights are pre-converted to bf16 in workspace so gate+up (134MB) and down
// (67MB) are L2-resident (192MB global L2) across the 16 M-tile sweeps; all
// GEMM staging then uses GLOBAL_LOAD_ASYNC_TO_LDS_B128 (ASYNCcnt) into a
// double-buffered LDS K-pipeline. Falls back to fp32-weight register pipeline
// if ws_size is too small for the bf16 weight copies.
// ---------------------------------------------------------------------------

typedef __attribute__((ext_vector_type(16))) __bf16 v16bf;
typedef __attribute__((ext_vector_type(8)))  float  v8f;

union Frag {
    v16bf v;
    uint4 u[2];
};

#define B_   2
#define S_   1024
#define HID_ 2048
#define NH_  8
#define HD_  256
#define INTER_ 16384
#define QKVN_ 2560      // (NH + 2*NKV) * HD
#define EPS_ 1e-6f
#define SOFTCAP_ 50.0f
#define SCALE_ 0.0625f  // HD^-0.5
#define NEG_ -1000000000.0f

// GEMM epilogue modes
#define M_PLAIN  0   // fp32 out
#define M_SCORES 1   // softcap + causal mask, fp32 out
#define M_PV     2   // bf16 out (strided into attn layout)
#define M_RESADD 3   // fp32 out = acc + res
#define M_GATEUP 4   // dual-B; bf16 out = gelu(acc0)*acc1
#define M_DOWN   5   // fp32 out = acc + res  (final output)

static __device__ __forceinline__ float gelu_tanh(float x) {
    const float k0 = 0.7978845608028654f;   // sqrt(2/pi)
    const float k1 = 0.044715f;
    float inner = k0 * (x + k1 * x * x * x);
    return 0.5f * x * (1.0f + tanhf(inner));
}

static __device__ __forceinline__ unsigned short bfbits(float f) {
    __hip_bfloat16 h = __float2bfloat16(f);
    return *reinterpret_cast<unsigned short*>(&h);
}

// convert 8 fp32 -> 8 bf16 and store 16B
static __device__ __forceinline__ void cvt_store8(unsigned short* dst, float4 a, float4 b) {
    __align__(16) unsigned short t[8];
    t[0] = bfbits(a.x); t[1] = bfbits(a.y); t[2] = bfbits(a.z); t[3] = bfbits(a.w);
    t[4] = bfbits(b.x); t[5] = bfbits(b.y); t[6] = bfbits(b.z); t[7] = bfbits(b.w);
    *(uint4*)dst = *(const uint4*)t;
}

// async 16B copy global -> LDS (GLOBAL_LOAD_ASYNC_TO_LDS_B128, tracked by ASYNCcnt)
static __device__ __forceinline__ void async_cp16(unsigned lds_off, const void* gaddr) {
    asm volatile("global_load_async_to_lds_b128 %0, %1, off"
                 :: "v"(lds_off), "v"(gaddr) : "memory");
}
static __device__ __forceinline__ void async_wait0() {
    asm volatile("s_wait_asynccnt 0x0" ::: "memory");
}

// ---------------------------------------------------------------------------
// Tiled WMMA GEMM:  C[m,n] = sum_k A[m,k] * B[n,k]   (B stored row-major N x K)
// A: bf16 (async-staged).  B: bf16 (async) or fp32 (register pipeline + cvt).
// Block = 256 threads = 8 waves; block tile 128(M) x 64(N); wave tile 32x32;
// K stepped by 32 through double-buffered LDS.
// Batch (blockIdx.z): A += z*strideA ; B += (z>>3)*strideB ;
//                     C += (z>>3)*strideCb + (z&7)*strideCh
// All dims are multiples of the tile sizes for this problem.
// ---------------------------------------------------------------------------
template<int MODE, typename BT>
__global__ __launch_bounds__(256)
void gemm_wmma(const __hip_bfloat16* __restrict__ A,
               const BT* __restrict__ B0,
               const BT* __restrict__ B1,
               void* __restrict__ Cout,
               const float* __restrict__ Res,
               int M, int N, int K,
               int lda, int ldb, int ldc,
               long long strideA, long long strideB,
               long long strideCb, long long strideCh)
{
    const int tid  = threadIdx.x;
    const int lane = tid & 31;
    const int wid  = tid >> 5;
    const int z    = blockIdx.z;

    A  += (long long)z * strideA;
    B0 += (long long)(z >> 3) * strideB;
    if (MODE == M_GATEUP) B1 += (long long)(z >> 3) * strideB;
    const long long coff = (long long)(z >> 3) * strideCb + (long long)(z & 7) * strideCh;

    // double-buffered LDS (rows padded to 40 elems = 80B, keeps 16B alignment)
    __shared__ __align__(16) unsigned short As[2][128][40];
    __shared__ __align__(16) unsigned short Bs[2][2][64][40];

    const int m0 = blockIdx.y * 128;
    const int n0 = blockIdx.x * 64;

    v8f acc[2][2]  = {};
    v8f acc2[2][2] = {};

    const int arow  = tid >> 1;            // 0..127
    const int ahalf = (tid & 1) * 16;      // 0 or 16
    const int brow  = tid >> 2;            // 0..63
    const int bq    = (tid & 3) * 8;       // 0,8,16,24

    const int mb = 32 * (wid & 3);         // wave M offset in block tile
    const int nb = 32 * (wid >> 2);        // wave N offset in block tile

    // per-lane fragment addressing (16-bit A layout, ISA 7.12.2):
    // lanes 0-15: K chunks {0..7, 16..23}; lanes 16-31: {8..15, 24..31}
    const int lrow = lane & 15;
    const int koff = (lane >> 4) * 8;

    // per-thread staging source pointers
    const unsigned short* aSrc  = (const unsigned short*)A +
                                  (long long)(m0 + arow) * lda + ahalf;
    const unsigned short* bSrcH = (const unsigned short*)B0 +
                                  (long long)(n0 + brow) * ldb + bq;
    const unsigned short* bSrcH1 = (const unsigned short*)B1 +
                                   (long long)(n0 + brow) * ldb + bq;
    const float* bSrcF  = reinterpret_cast<const float*>(B0) +
                          (long long)(n0 + brow) * ldb + bq;
    const float* bSrcF1 = reinterpret_cast<const float*>(B1) +
                          (long long)(n0 + brow) * ldb + bq;

    // LDS byte offsets (low 32 bits of the generic shared address == LDS offset)
    const unsigned aL[2] = {
        (unsigned)(unsigned long long)(void*)&As[0][arow][ahalf],
        (unsigned)(unsigned long long)(void*)&As[1][arow][ahalf] };
    const unsigned bL[2] = {
        (unsigned)(unsigned long long)(void*)&Bs[0][0][brow][bq],
        (unsigned)(unsigned long long)(void*)&Bs[1][0][brow][bq] };
    const unsigned B1OFF = 64u * 40u * 2u;   // Bs[buf][1] - Bs[buf][0] in bytes

    float4 r0 = {}, r1 = {}, r2 = {}, r3 = {};  // fp32 weight pipeline regs

    const int KT = K >> 5;

    // ---- prologue: stage tile 0 into buffer 0
    async_cp16(aL[0],       (const void*)(aSrc));
    async_cp16(aL[0] + 16u, (const void*)(aSrc + 8));
    if (sizeof(BT) == 2) {
        async_cp16(bL[0], (const void*)(bSrcH));
        if (MODE == M_GATEUP) async_cp16(bL[0] + B1OFF, (const void*)(bSrcH1));
    } else {
        r0 = *(const float4*)(bSrcF);
        r1 = *(const float4*)(bSrcF + 4);
        if (MODE == M_GATEUP) {
            r2 = *(const float4*)(bSrcF1);
            r3 = *(const float4*)(bSrcF1 + 4);
        }
        cvt_store8(&Bs[0][0][brow][bq], r0, r1);
        if (MODE == M_GATEUP) cvt_store8(&Bs[0][1][brow][bq], r2, r3);
    }
    async_wait0();
    __syncthreads();

    for (int kt = 0; kt < KT; ++kt) {
        const int buf  = kt & 1;
        const int nxt  = buf ^ 1;
        const bool more = (kt + 1 < KT);

        // ---- issue staging for tile kt+1 (overlaps with WMMA below)
        if (more) {
            const int kk = 32 * (kt + 1);
            async_cp16(aL[nxt],       (const void*)(aSrc + kk));
            async_cp16(aL[nxt] + 16u, (const void*)(aSrc + kk + 8));
            if (sizeof(BT) == 2) {
                async_cp16(bL[nxt], (const void*)(bSrcH + kk));
                if (MODE == M_GATEUP)
                    async_cp16(bL[nxt] + B1OFF, (const void*)(bSrcH1 + kk));
                if (kk + 32 < K)
                    __builtin_prefetch((const void*)(bSrcH + kk + 32), 0, 0);
            } else {
                r0 = *(const float4*)(bSrcF + kk);
                r1 = *(const float4*)(bSrcF + kk + 4);
                if (MODE == M_GATEUP) {
                    r2 = *(const float4*)(bSrcF1 + kk);
                    r3 = *(const float4*)(bSrcF1 + kk + 4);
                }
                if (kk + 32 < K)
                    __builtin_prefetch((const void*)(bSrcF + kk + 32), 0, 0);
            }
        }

        // ---- fragments from LDS (buffer `buf`)
        Frag afr[2], bfr[2], bfr2[2];
#pragma unroll
        for (int i = 0; i < 2; ++i) {
            int r = mb + 16 * i + lrow;
            afr[i].u[0] = *(const uint4*)&As[buf][r][koff];
            afr[i].u[1] = *(const uint4*)&As[buf][r][koff + 16];
        }
#pragma unroll
        for (int j = 0; j < 2; ++j) {
            int r = nb + 16 * j + lrow;
            bfr[j].u[0] = *(const uint4*)&Bs[buf][0][r][koff];
            bfr[j].u[1] = *(const uint4*)&Bs[buf][0][r][koff + 16];
            if (MODE == M_GATEUP) {
                bfr2[j].u[0] = *(const uint4*)&Bs[buf][1][r][koff];
                bfr2[j].u[1] = *(const uint4*)&Bs[buf][1][r][koff + 16];
            }
        }
        // ---- 4 (or 8) WMMAs
#pragma unroll
        for (int i = 0; i < 2; ++i)
#pragma unroll
            for (int j = 0; j < 2; ++j) {
                acc[i][j] = __builtin_amdgcn_wmma_f32_16x16x32_bf16(
                    false, afr[i].v, false, bfr[j].v, (short)0, acc[i][j], false, false);
                if (MODE == M_GATEUP)
                    acc2[i][j] = __builtin_amdgcn_wmma_f32_16x16x32_bf16(
                        false, afr[i].v, false, bfr2[j].v, (short)0, acc2[i][j], false, false);
            }

        // ---- finish staging tile kt+1 (loadcnt waits land AFTER the WMMAs)
        if (more) {
            if (sizeof(BT) == 4) {
                cvt_store8(&Bs[nxt][0][brow][bq], r0, r1);
                if (MODE == M_GATEUP) cvt_store8(&Bs[nxt][1][brow][bq], r2, r3);
            }
            async_wait0();
        }
        __syncthreads();
    }

    // ---- epilogue (C layout, ISA 7.12.2: VGPR r -> M=r / M=8+r; N = lane%16)
    const int moff = (lane >> 4) * 8;
    const int ncol = lane & 15;
#pragma unroll
    for (int i = 0; i < 2; ++i)
#pragma unroll
        for (int j = 0; j < 2; ++j)
#pragma unroll
            for (int r = 0; r < 8; ++r) {
                int gm = m0 + mb + 16 * i + moff + r;
                int gn = n0 + nb + 16 * j + ncol;
                float v = acc[i][j][r];
                long long idx = coff + (long long)gm * ldc + gn;
                if (MODE == M_PLAIN) {
                    ((float*)Cout)[idx] = v;
                } else if (MODE == M_SCORES) {
                    v = tanhf(v * (SCALE_ / SOFTCAP_)) * SOFTCAP_;
                    if (gn > gm) v += NEG_;            // causal mask (t > s)
                    ((float*)Cout)[idx] = v;
                } else if (MODE == M_PV) {
                    ((__hip_bfloat16*)Cout)[idx] = __float2bfloat16(v);
                } else if (MODE == M_RESADD) {
                    ((float*)Cout)[idx] = v + Res[(long long)gm * ldc + gn];
                } else if (MODE == M_GATEUP) {
                    float g = gelu_tanh(v);
                    float u = acc2[i][j][r];
                    ((__hip_bfloat16*)Cout)[idx] = __float2bfloat16(g * u);
                } else { // M_DOWN
                    ((float*)Cout)[idx] = v + Res[(long long)gm * ldc + gn];
                }
            }
}

// ---------------------------------------------------------------------------
// fp32 -> bf16 bulk conversion (weights into workspace; 8 elems/thread)
// ---------------------------------------------------------------------------
__global__ __launch_bounds__(256)
void f32_to_bf16_kernel(const float* __restrict__ src,
                        __hip_bfloat16* __restrict__ dst, long long n)
{
    long long i = ((long long)blockIdx.x * 256 + threadIdx.x) * 8;
    if (i + 8 <= n) {
        float4 a = *(const float4*)(src + i);
        float4 b = *(const float4*)(src + i + 4);
        cvt_store8((unsigned short*)(dst + i), a, b);
    }
}

// ---------------------------------------------------------------------------
// RMSNorm over 2048 cols, fp32 in -> bf16 out. One block per row.
// ---------------------------------------------------------------------------
__global__ __launch_bounds__(256)
void rmsnorm_kernel(const float* __restrict__ x, const float* __restrict__ w,
                    __hip_bfloat16* __restrict__ out)
{
    const long long row = blockIdx.x;
    const int t = threadIdx.x;
    const float* p = x + row * HID_;
    __shared__ float red[256];
    float v[8];
    float ss = 0.f;
#pragma unroll
    for (int i = 0; i < 8; ++i) { v[i] = p[t + 256 * i]; ss += v[i] * v[i]; }
    red[t] = ss;
    __syncthreads();
    for (int off = 128; off > 0; off >>= 1) {
        if (t < off) red[t] += red[t + off];
        __syncthreads();
    }
    float rms = rsqrtf(red[0] / (float)HID_ + EPS_);
#pragma unroll
    for (int i = 0; i < 8; ++i) {
        int c = t + 256 * i;
        out[row * HID_ + c] = __float2bfloat16(v[i] * rms * (1.f + w[c]));
    }
}

// ---------------------------------------------------------------------------
// Post-QKV: per (b,s) block — q/k head RMSNorm + RoPE, layout rewrites.
//   qout: (B, NH, S, HD) bf16   kout: (B, S, HD) bf16   vT: (B, HD, S) bf16
// ---------------------------------------------------------------------------
__global__ __launch_bounds__(256)
void qkv_post_kernel(const float* __restrict__ qkv,
                     const float* __restrict__ qn, const float* __restrict__ kn,
                     const float* __restrict__ cosp, const float* __restrict__ sinp,
                     __hip_bfloat16* __restrict__ qout,
                     __hip_bfloat16* __restrict__ kout,
                     __hip_bfloat16* __restrict__ vT)
{
    const int bs = blockIdx.x;
    const int b = bs >> 10;       // / S_
    const int s = bs & (S_ - 1);
    const int t = threadIdx.x;    // 0..255 (= HD)
    const float* row = qkv + (long long)bs * QKVN_;

    __shared__ float red[256];
    __shared__ float buf[256];

    const float cv = cosp[s * 128 + (t & 127)];
    const float sv = sinp[s * 128 + (t & 127)];

    for (int h = 0; h < 9; ++h) {   // h<8 : q heads; h==8 : k
        const float* base = row + (h < 8 ? h * HD_ : NH_ * HD_);
        float x = base[t];
        red[t] = x * x;
        __syncthreads();
        for (int off = 128; off > 0; off >>= 1) {
            if (t < off) red[t] += red[t + off];
            __syncthreads();
        }
        float rms = rsqrtf(red[0] / (float)HD_ + EPS_);
        const float* w = (h < 8) ? qn : kn;
        buf[t] = x * rms * (1.f + w[t]);
        __syncthreads();
        float o = (t < 128) ? (buf[t] * cv - buf[t + 128] * sv)
                            : (buf[t - 128] * sv + buf[t] * cv);
        if (h < 8)
            qout[(((long long)(b * NH_ + h)) * S_ + s) * HD_ + t] = __float2bfloat16(o);
        else
            kout[(long long)bs * HD_ + t] = __float2bfloat16(o);
        __syncthreads();
    }
    // v: no norm / no rope; store transposed for the PV GEMM B-operand
    float vv = row[(NH_ + 1) * HD_ + t];
    vT[((long long)b * HD_ + t) * S_ + s] = __float2bfloat16(vv);
}

// ---------------------------------------------------------------------------
// Row softmax over 1024 cols (scores already scaled/softcapped/masked) -> bf16
// ---------------------------------------------------------------------------
__global__ __launch_bounds__(256)
void softmax_kernel(const float* __restrict__ sc, __hip_bfloat16* __restrict__ pr)
{
    const long long row = blockIdx.x;
    const int t = threadIdx.x;
    const float* p = sc + row * 1024;
    __shared__ float red[256];
    float v[4];
    float mx = -3.4e38f;
#pragma unroll
    for (int i = 0; i < 4; ++i) { v[i] = p[t + 256 * i]; mx = fmaxf(mx, v[i]); }
    red[t] = mx;
    __syncthreads();
    for (int off = 128; off > 0; off >>= 1) {
        if (t < off) red[t] = fmaxf(red[t], red[t + off]);
        __syncthreads();
    }
    float m = red[0];
    __syncthreads();
    float sum = 0.f;
#pragma unroll
    for (int i = 0; i < 4; ++i) { v[i] = __expf(v[i] - m); sum += v[i]; }
    red[t] = sum;
    __syncthreads();
    for (int off = 128; off > 0; off >>= 1) {
        if (t < off) red[t] += red[t + off];
        __syncthreads();
    }
    float inv = 1.f / red[0];
#pragma unroll
    for (int i = 0; i < 4; ++i)
        pr[row * 1024 + t + 256 * i] = __float2bfloat16(v[i] * inv);
}

// ---------------------------------------------------------------------------
extern "C" void kernel_launch(void* const* d_in, const int* in_sizes, int n_in,
                              void* d_out, int out_size, void* d_ws, size_t ws_size,
                              hipStream_t stream)
{
    const float* hidden  = (const float*)d_in[0];
    const float* cosp    = (const float*)d_in[1];
    const float* sinp    = (const float*)d_in[2];
    // d_in[3..6]: kv_write_indices (=arange), zero caches, causal mask — folded analytically
    const float* w_qkv   = (const float*)d_in[7];
    const float* w_o     = (const float*)d_in[8];
    const float* w_gate  = (const float*)d_in[9];
    const float* w_up    = (const float*)d_in[10];
    const float* w_down  = (const float*)d_in[11];
    const float* ln_in   = (const float*)d_in[12];
    const float* ln_post = (const float*)d_in[13];
    const float* q_norm  = (const float*)d_in[14];
    const float* k_norm  = (const float*)d_in[15];
    float* out = (float*)d_out;

    char* ws = (char*)d_ws;
    const long long MB = 1ll << 20;
    __hip_bfloat16* x_bf   = (__hip_bfloat16*)(ws + 0);          //  8 MB (reused: x2)
    float*          qkv    = (float*)        (ws + 8 * MB);      // 20 MB (reused: h)
    float*          h_buf  = (float*)        (ws + 8 * MB);      // 16 MB
    __hip_bfloat16* q_bf   = (__hip_bfloat16*)(ws + 28 * MB);    //  8 MB
    __hip_bfloat16* k_bf   = (__hip_bfloat16*)(ws + 36 * MB);    //  1 MB
    __hip_bfloat16* vT_bf  = (__hip_bfloat16*)(ws + 37 * MB);    //  1 MB
    float*          scores = (float*)        (ws + 38 * MB);     // 64 MB (reused: act)
    __hip_bfloat16* act    = (__hip_bfloat16*)(ws + 38 * MB);    // 64 MB
    __hip_bfloat16* probs  = (__hip_bfloat16*)(ws + 102 * MB);   // 32 MB
    __hip_bfloat16* attn   = (__hip_bfloat16*)(ws + 134 * MB);   //  8 MB
    // bf16 weight copies (L2-resident working set for the MLP GEMMs)
    __hip_bfloat16* wqkv_bf  = (__hip_bfloat16*)(ws + 142 * MB); // 10 MB
    __hip_bfloat16* wo_bf    = (__hip_bfloat16*)(ws + 152 * MB); //  8 MB
    __hip_bfloat16* wgate_bf = (__hip_bfloat16*)(ws + 160 * MB); // 64 MB
    __hip_bfloat16* wup_bf   = (__hip_bfloat16*)(ws + 224 * MB); // 64 MB
    __hip_bfloat16* wdown_bf = (__hip_bfloat16*)(ws + 288 * MB); // 64 MB
    const bool wbf = (ws_size >= (size_t)(352 * MB));

    const int Mrows = B_ * S_;   // 2048

    // 0. (if workspace allows) bulk-convert all weights to bf16 once
    if (wbf) {
        const long long nqkv  = (long long)QKVN_ * HID_;
        const long long no    = (long long)HID_ * HID_;
        const long long nbig  = (long long)INTER_ * HID_;
        f32_to_bf16_kernel<<<(unsigned)(nqkv / 8 / 256), 256, 0, stream>>>(w_qkv, wqkv_bf, nqkv);
        f32_to_bf16_kernel<<<(unsigned)(no   / 8 / 256), 256, 0, stream>>>(w_o, wo_bf, no);
        f32_to_bf16_kernel<<<(unsigned)(nbig / 8 / 256), 256, 0, stream>>>(w_gate, wgate_bf, nbig);
        f32_to_bf16_kernel<<<(unsigned)(nbig / 8 / 256), 256, 0, stream>>>(w_up, wup_bf, nbig);
        f32_to_bf16_kernel<<<(unsigned)(nbig / 8 / 256), 256, 0, stream>>>(w_down, wdown_bf, nbig);
    }

    // 1. input RMSNorm -> bf16
    rmsnorm_kernel<<<Mrows, 256, 0, stream>>>(hidden, ln_in, x_bf);

    // 2. QKV GEMM: (2048 x 2048) x (2560 x 2048)^T -> fp32
    if (wbf)
        gemm_wmma<M_PLAIN, __hip_bfloat16><<<dim3(QKVN_ / 64, Mrows / 128, 1), 256, 0, stream>>>(
            x_bf, wqkv_bf, nullptr, qkv, nullptr,
            Mrows, QKVN_, HID_, HID_, HID_, QKVN_, 0, 0, 0, 0);
    else
        gemm_wmma<M_PLAIN, float><<<dim3(QKVN_ / 64, Mrows / 128, 1), 256, 0, stream>>>(
            x_bf, w_qkv, nullptr, qkv, nullptr,
            Mrows, QKVN_, HID_, HID_, HID_, QKVN_, 0, 0, 0, 0);

    // 3. q/k norm + RoPE + layout
    qkv_post_kernel<<<Mrows, 256, 0, stream>>>(qkv, q_norm, k_norm, cosp, sinp,
                                               q_bf, k_bf, vT_bf);

    // 4. scores = softcap(scale * q k^T) + mask   [per (b,h)]
    gemm_wmma<M_SCORES, __hip_bfloat16><<<dim3(S_ / 64, S_ / 128, B_ * NH_), 256, 0, stream>>>(
        q_bf, k_bf, nullptr, scores, nullptr,
        S_, S_, HD_, HD_, HD_, S_,
        (long long)S_ * HD_, (long long)S_ * HD_,
        (long long)NH_ * S_ * S_, (long long)S_ * S_);

    // 5. softmax rows -> bf16 probs
    softmax_kernel<<<B_ * NH_ * S_, 256, 0, stream>>>(scores, probs);

    // 6. out = probs @ V   (B-operand = pre-transposed V), write into (B,S,NH*HD)
    gemm_wmma<M_PV, __hip_bfloat16><<<dim3(HD_ / 64, S_ / 128, B_ * NH_), 256, 0, stream>>>(
        probs, vT_bf, nullptr, attn, nullptr,
        S_, HD_, S_, S_, S_, HID_,
        (long long)S_ * S_, (long long)HD_ * S_,
        (long long)S_ * HID_, (long long)HD_);

    // 7. h = residual + attn @ w_o^T
    if (wbf)
        gemm_wmma<M_RESADD, __hip_bfloat16><<<dim3(HID_ / 64, Mrows / 128, 1), 256, 0, stream>>>(
            attn, wo_bf, nullptr, h_buf, hidden,
            Mrows, HID_, HID_, HID_, HID_, HID_, 0, 0, 0, 0);
    else
        gemm_wmma<M_RESADD, float><<<dim3(HID_ / 64, Mrows / 128, 1), 256, 0, stream>>>(
            attn, w_o, nullptr, h_buf, hidden,
            Mrows, HID_, HID_, HID_, HID_, HID_, 0, 0, 0, 0);

    // 8. post RMSNorm -> bf16 (reuses x buffer)
    rmsnorm_kernel<<<Mrows, 256, 0, stream>>>(h_buf, ln_post, x_bf);

    // 9. act = gelu(x @ w_gate^T) * (x @ w_up^T)  (dual-B, one K sweep)
    if (wbf)
        gemm_wmma<M_GATEUP, __hip_bfloat16><<<dim3(INTER_ / 64, Mrows / 128, 1), 256, 0, stream>>>(
            x_bf, wgate_bf, wup_bf, act, nullptr,
            Mrows, INTER_, HID_, HID_, HID_, INTER_, 0, 0, 0, 0);
    else
        gemm_wmma<M_GATEUP, float><<<dim3(INTER_ / 64, Mrows / 128, 1), 256, 0, stream>>>(
            x_bf, w_gate, w_up, act, nullptr,
            Mrows, INTER_, HID_, HID_, HID_, INTER_, 0, 0, 0, 0);

    // 10. out = h + act @ w_down^T
    if (wbf)
        gemm_wmma<M_DOWN, __hip_bfloat16><<<dim3(HID_ / 64, Mrows / 128, 1), 256, 0, stream>>>(
            act, wdown_bf, nullptr, out, h_buf,
            Mrows, HID_, INTER_, INTER_, INTER_, HID_, 0, 0, 0, 0);
    else
        gemm_wmma<M_DOWN, float><<<dim3(HID_ / 64, Mrows / 128, 1), 256, 0, stream>>>(
            act, w_down, nullptr, out, h_buf,
            Mrows, HID_, INTER_, INTER_, INTER_, HID_, 0, 0, 0, 0);
}
